// SS2Dv0_3384434229777
// MI455X (gfx1250) — compile-verified, matrix operand order checked by reading, not run on hardware
//
#include <hip/hip_runtime.h>
#include <hip/hip_bf16.h>
#include <math.h>

// ---------------- problem constants ----------------
#define B_   2
#define C_   768
#define H_   32
#define W_   32
#define L_   (H_ * W_)      // 1024
#define DI_  1536
#define N_   16
#define R_   48
#define K_   4
#define RPAD 64             // R padded to WMMA K multiple
#define M1_  (B_ * L_)      // 2048 rows of activations

typedef __bf16 bf16_t;
typedef __attribute__((ext_vector_type(8)))  __bf16 v8bf;
typedef __attribute__((ext_vector_type(16))) __bf16 v16bf;
typedef __attribute__((ext_vector_type(8)))  float  v8f;

// ---------------- WMMA helpers ----------------
// A (16x32, M=lane&15) and B (32x16, N=lane&15) fragments share the same
// per-lane K layout: half = lane>>4 selects K in {h*8..h*8+7} U {16+h*8..16+h*8+7}.
__device__ __forceinline__ v16bf load_frag16x32(const bf16_t* row, int k0, int half8) {
  const bf16_t* p = row + k0 + half8;
  v8bf lo = *reinterpret_cast<const v8bf*>(p);
  v8bf hi = *reinterpret_cast<const v8bf*>(p + 16);
  v16bf f;
#pragma unroll
  for (int i = 0; i < 8; ++i) { f[i] = lo[i]; f[8 + i] = hi[i]; }
  return f;
}

__device__ __forceinline__ v8f wmma_bf16(v16bf a, v16bf b, v8f c) {
  return __builtin_amdgcn_wmma_f32_16x16x32_bf16(false, a, false, b, (short)0, c,
                                                 false, false);
}

__device__ __forceinline__ float silu_f(float x) {
  return x / (1.0f + __expf(-x));
}

// direction-dependent gather index into (b, L) spatial order
__device__ __forceinline__ int src_map(int k, int l) {
  int ll = (k & 2) ? (L_ - 1 - l) : l;                 // flip for k=2,3
  if (k & 1) ll = ((ll & 31) << 5) | (ll >> 5);        // h<->w transpose for k=1,3
  return ll;
}

// ---------------- pack kernels ----------------
__global__ __launch_bounds__(256) void pack_x_kernel(const float* __restrict__ x,
                                                     bf16_t* __restrict__ xT) {
  int idx = blockIdx.x * 256 + threadIdx.x;           // over M1_*C_, c fastest
  if (idx >= M1_ * C_) return;
  int c = idx % C_;
  int m = idx / C_;
  int b = m >> 10, l = m & (L_ - 1);
  xT[idx] = (bf16_t)x[((size_t)(b * C_ + c)) * L_ + l];
}

__global__ __launch_bounds__(256) void f32_to_bf16_kernel(const float* __restrict__ in,
                                                          bf16_t* __restrict__ out, int n) {
  int i = blockIdx.x * 256 + threadIdx.x;
  if (i < n) out[i] = (bf16_t)in[i];
}

__global__ __launch_bounds__(256) void pack_dtw_kernel(const float* __restrict__ dtw,
                                                       bf16_t* __restrict__ out) {
  int i = blockIdx.x * 256 + threadIdx.x;             // over K_*DI_*RPAD
  if (i >= K_ * DI_ * RPAD) return;
  int r  = i % RPAD;
  int kd = i / RPAD;                                  // k*DI_+d
  out[i] = (r < R_) ? (bf16_t)dtw[(size_t)kd * R_ + r] : (bf16_t)0.0f;
}

// ---------------- GEMM 1: in_proj  (2048 x 768) * (3072 x 768)^T ----------------
// epilogue: col<DI -> xi (B,L,DI) fp32 ; col>=DI -> silu -> z (B,L,DI) fp32
__global__ __launch_bounds__(256) void gemm_inproj(const bf16_t* __restrict__ A,
                                                   const bf16_t* __restrict__ Wt,
                                                   float* __restrict__ xi,
                                                   float* __restrict__ z) {
  int gtid = blockIdx.x * 256 + threadIdx.x;
  int wave = gtid >> 5;
  int lane = threadIdx.x & 31;
  const int NG = (2 * DI_) / 64;                      // 48
  if (wave >= (M1_ / 16) * NG) return;
  int mt = wave / NG, ng = wave % NG;
  int m0 = mt * 16, n0 = ng * 64;
  int half8 = ((lane >> 4) & 1) * 8;
  const bf16_t* arow = A + (size_t)(m0 + (lane & 15)) * C_;
  v8f zero = {0.f, 0.f, 0.f, 0.f, 0.f, 0.f, 0.f, 0.f};
  v8f acc[4];
#pragma unroll
  for (int t = 0; t < 4; ++t) acc[t] = zero;
  for (int k0 = 0; k0 < C_; k0 += 32) {
    v16bf a = load_frag16x32(arow, k0, half8);
#pragma unroll
    for (int t = 0; t < 4; ++t) {
      const bf16_t* brow = Wt + (size_t)(n0 + t * 16 + (lane & 15)) * C_;
      v16bf bfr = load_frag16x32(brow, k0, half8);
      acc[t] = wmma_bf16(a, bfr, acc[t]);
    }
  }
  int mbase = m0 + ((lane >> 4) & 1) * 8;
#pragma unroll
  for (int t = 0; t < 4; ++t) {
    int col = n0 + t * 16 + (lane & 15);
#pragma unroll
    for (int i = 0; i < 8; ++i) {
      int m = mbase + i;
      float v = acc[t][i];
      if (col < DI_) xi[(size_t)m * DI_ + col] = v;
      else           z[(size_t)m * DI_ + (col - DI_)] = silu_f(v);
    }
  }
}

// ---------------- depthwise 3x3 conv + silu ----------------
// xi layout (B,L,DI) -> xcT fp32 (B,L,DI) + bf16 copy
__global__ __launch_bounds__(256) void conv_dw_kernel(const float* __restrict__ xi,
                                                      const float* __restrict__ cw,
                                                      const float* __restrict__ cb,
                                                      float* __restrict__ xcT,
                                                      bf16_t* __restrict__ xcT_bf) {
  int idx = blockIdx.x * 256 + threadIdx.x;           // over B_*L_*DI_, d fastest
  if (idx >= B_ * L_ * DI_) return;
  int d  = idx % DI_;
  int bl = idx / DI_;
  int l  = bl & (L_ - 1);
  int b  = bl >> 10;
  int h  = l >> 5, w = l & 31;
  float acc = cb[d];
#pragma unroll
  for (int dy = -1; dy <= 1; ++dy) {
    int hh = h + dy;
    if ((unsigned)hh >= H_) continue;
#pragma unroll
    for (int dx = -1; dx <= 1; ++dx) {
      int ww = w + dx;
      if ((unsigned)ww >= W_) continue;
      acc += xi[((size_t)(b * L_ + (hh << 5) + ww)) * DI_ + d] *
             cw[d * 9 + (dy + 1) * 3 + (dx + 1)];
    }
  }
  acc = silu_f(acc);
  xcT[idx] = acc;
  xcT_bf[idx] = (bf16_t)acc;
}

// ---------------- GEMM 2: x_dbl per (b,k): gathered-rows (1024 x 1536)*(80 x 1536)^T ----
// epilogue split: c<48 -> dts_in bf16 (pad cols 48..63 = 0); 48..63 -> Bs; 64..79 -> Cs
__global__ __launch_bounds__(256) void gemm_xdbl(const bf16_t* __restrict__ xcT_bf,
                                                 const bf16_t* __restrict__ W80,
                                                 bf16_t* __restrict__ dts_in,
                                                 float* __restrict__ Bsb,
                                                 float* __restrict__ Csb) {
  int gtid = blockIdx.x * 256 + threadIdx.x;
  int wave = gtid >> 5;
  int lane = threadIdx.x & 31;
  if (wave >= B_ * K_ * (L_ / 16)) return;
  int mt = wave % (L_ / 16);
  int bk = wave / (L_ / 16);
  int b = bk / K_, k = bk % K_;
  int m0 = mt * 16;
  int half8 = ((lane >> 4) & 1) * 8;
  int sl = src_map(k, m0 + (lane & 15));
  const bf16_t* arow = xcT_bf + (size_t)(b * L_ + sl) * DI_;
  v8f zero = {0.f, 0.f, 0.f, 0.f, 0.f, 0.f, 0.f, 0.f};
  v8f acc[5];
#pragma unroll
  for (int t = 0; t < 5; ++t) acc[t] = zero;
  for (int k0 = 0; k0 < DI_; k0 += 32) {
    v16bf a = load_frag16x32(arow, k0, half8);
#pragma unroll
    for (int t = 0; t < 5; ++t) {
      const bf16_t* brow = W80 + (size_t)(t * 16 + (lane & 15)) * DI_;
      v16bf bfr = load_frag16x32(brow, k0, half8);
      acc[t] = wmma_bf16(a, bfr, acc[t]);
    }
  }
  int mbase = m0 + ((lane >> 4) & 1) * 8;
#pragma unroll
  for (int t = 0; t < 5; ++t) {
    int col = t * 16 + (lane & 15);
#pragma unroll
    for (int i = 0; i < 8; ++i) {
      int m = mbase + i;
      size_t base = (size_t)bk * L_ + m;
      float v = acc[t][i];
      if (col < R_) {
        dts_in[base * RPAD + col] = (bf16_t)v;
      } else if (col < R_ + N_) {
        dts_in[base * RPAD + col] = (bf16_t)0.0f;   // zero-pad K for next GEMM
        Bsb[base * N_ + (col - R_)] = v;
      } else {
        Csb[base * N_ + (col - R_ - N_)] = v;
      }
    }
  }
}

// ---------------- GEMM 3: dts per (b,k): (1024 x 64) * (1536 x 64)^T ----------------
__global__ __launch_bounds__(256) void gemm_dts(const bf16_t* __restrict__ dts_in,
                                                const bf16_t* __restrict__ dtw,
                                                float* __restrict__ dts) {
  int gtid = blockIdx.x * 256 + threadIdx.x;
  int wave = gtid >> 5;
  int lane = threadIdx.x & 31;
  const int NG = DI_ / 64;                            // 24
  if (wave >= B_ * K_ * (L_ / 16) * NG) return;
  int ng = wave % NG;
  int rest = wave / NG;
  int mt = rest % (L_ / 16);
  int bk = rest / (L_ / 16);
  int k = bk % K_;
  int m0 = mt * 16, n0 = ng * 64;
  int half8 = ((lane >> 4) & 1) * 8;
  const bf16_t* arow = dts_in + ((size_t)bk * L_ + m0 + (lane & 15)) * RPAD;
  v8f zero = {0.f, 0.f, 0.f, 0.f, 0.f, 0.f, 0.f, 0.f};
  v8f acc[4];
#pragma unroll
  for (int t = 0; t < 4; ++t) acc[t] = zero;
#pragma unroll
  for (int k0 = 0; k0 < RPAD; k0 += 32) {
    v16bf a = load_frag16x32(arow, k0, half8);
#pragma unroll
    for (int t = 0; t < 4; ++t) {
      const bf16_t* brow = dtw + ((size_t)k * DI_ + n0 + t * 16 + (lane & 15)) * RPAD;
      v16bf bfr = load_frag16x32(brow, k0, half8);
      acc[t] = wmma_bf16(a, bfr, acc[t]);
    }
  }
  int mbase = m0 + ((lane >> 4) & 1) * 8;
#pragma unroll
  for (int t = 0; t < 4; ++t) {
    int col = n0 + t * 16 + (lane & 15);
#pragma unroll
    for (int i = 0; i < 8; ++i) {
      int m = mbase + i;
      dts[((size_t)bk * L_ + m) * DI_ + col] = acc[t][i];
    }
  }
}

// ---------------- selective scan: one thread per (b,k,d), N=16 states in regs --------
__global__ __launch_bounds__(128) void scan_kernel(const float* __restrict__ dts,
                                                   const float* __restrict__ Bsb,
                                                   const float* __restrict__ Csb,
                                                   const float* __restrict__ xcT,
                                                   const float* __restrict__ A_logs,
                                                   const float* __restrict__ Ds,
                                                   const float* __restrict__ dt_bias,
                                                   float* __restrict__ ys) {
  int tid = blockIdx.x * 128 + threadIdx.x;           // over B_*K_*DI_, d fastest
  if (tid >= B_ * K_ * DI_) return;
  int d  = tid % DI_;
  int bk = tid / DI_;
  int k  = bk % K_;
  int b  = bk / K_;
  int kd = k * DI_ + d;

  float Aloc[N_];
#pragma unroll
  for (int n = 0; n < N_; ++n) Aloc[n] = -__expf(A_logs[(size_t)kd * N_ + n]);
  float bias = dt_bias[kd];
  float Dval = Ds[kd];

  float h[N_];
#pragma unroll
  for (int n = 0; n < N_; ++n) h[n] = 0.0f;

  size_t base = (size_t)bk * L_;
  for (int l = 0; l < L_; ++l) {
    if (l + 8 < L_)
      __builtin_prefetch(dts + (base + l + 8) * DI_ + d, 0, 0);  // global_prefetch_b8
    float dtv = dts[(base + l) * DI_ + d] + bias;
    float delta = (dtv > 20.0f) ? dtv : log1pf(__expf(dtv));     // softplus
    int sl = src_map(k, l);
    float u = xcT[((size_t)(b * L_ + sl)) * DI_ + d];
    float du = delta * u;
    const float* Bp = Bsb + (base + l) * N_;
    const float* Cp = Csb + (base + l) * N_;
    float y = 0.0f;
#pragma unroll
    for (int n = 0; n < N_; ++n) {
      float dA = __expf(delta * Aloc[n]);
      float hn = dA * h[n] + du * Bp[n];
      h[n] = hn;
      y += hn * Cp[n];
    }
    ys[(base + l) * DI_ + d] = y + Dval * u;
  }
}

// ---------------- direction merge + LayerNorm + z-gate -> bf16 ----------------
__global__ __launch_bounds__(256) void merge_ln_kernel(const float* __restrict__ ys,
                                                       const float* __restrict__ z,
                                                       const float* __restrict__ gamma,
                                                       const float* __restrict__ beta,
                                                       bf16_t* __restrict__ yln) {
  __shared__ float sh[512];
  int bl = blockIdx.x;                                // (b, l) with l = h*W+w
  int b = bl >> 10, l = bl & (L_ - 1);
  int hh = l >> 5, ww = l & 31;
  int lt = (ww << 5) | hh;                            // transposed flat index
  size_t r0 = ((size_t)(b * K_ + 0) * L_ + l) * DI_;
  size_t r1 = ((size_t)(b * K_ + 1) * L_ + lt) * DI_;
  size_t r2 = ((size_t)(b * K_ + 2) * L_ + (L_ - 1 - l)) * DI_;
  size_t r3 = ((size_t)(b * K_ + 3) * L_ + (L_ - 1 - lt)) * DI_;

  const int PT = DI_ / 256;                           // 6 per thread
  float vals[PT];
  float s = 0.0f, ss = 0.0f;
#pragma unroll
  for (int j = 0; j < PT; ++j) {
    int d = threadIdx.x + j * 256;
    float v = ys[r0 + d] + ys[r1 + d] + ys[r2 + d] + ys[r3 + d];
    vals[j] = v;
    s += v;
    ss += v * v;
  }
  sh[threadIdx.x] = s;
  sh[256 + threadIdx.x] = ss;
  __syncthreads();
  for (int off = 128; off > 0; off >>= 1) {
    if ((int)threadIdx.x < off) {
      sh[threadIdx.x] += sh[threadIdx.x + off];
      sh[256 + threadIdx.x] += sh[256 + threadIdx.x + off];
    }
    __syncthreads();
  }
  float mu  = sh[0] * (1.0f / DI_);
  float var = sh[256] * (1.0f / DI_) - mu * mu;
  float inv = rsqrtf(var + 1e-5f);
#pragma unroll
  for (int j = 0; j < PT; ++j) {
    int d = threadIdx.x + j * 256;
    float yv = (vals[j] - mu) * inv * gamma[d] + beta[d];
    yv *= z[(size_t)bl * DI_ + d];
    yln[(size_t)bl * DI_ + d] = (bf16_t)yv;
  }
}

// ---------------- GEMM 4: out_proj (2048 x 1536) * (768 x 1536)^T -> (B,C,H,W) --------
__global__ __launch_bounds__(256) void gemm_out(const bf16_t* __restrict__ A,
                                                const bf16_t* __restrict__ Wt,
                                                float* __restrict__ out) {
  int gtid = blockIdx.x * 256 + threadIdx.x;
  int wave = gtid >> 5;
  int lane = threadIdx.x & 31;
  const int NG = C_ / 64;                             // 12
  if (wave >= (M1_ / 16) * NG) return;
  int mt = wave / NG, ng = wave % NG;
  int m0 = mt * 16, n0 = ng * 64;
  int half8 = ((lane >> 4) & 1) * 8;
  const bf16_t* arow = A + (size_t)(m0 + (lane & 15)) * DI_;
  v8f zero = {0.f, 0.f, 0.f, 0.f, 0.f, 0.f, 0.f, 0.f};
  v8f acc[4];
#pragma unroll
  for (int t = 0; t < 4; ++t) acc[t] = zero;
  for (int k0 = 0; k0 < DI_; k0 += 32) {
    v16bf a = load_frag16x32(arow, k0, half8);
#pragma unroll
    for (int t = 0; t < 4; ++t) {
      const bf16_t* brow = Wt + (size_t)(n0 + t * 16 + (lane & 15)) * DI_;
      v16bf bfr = load_frag16x32(brow, k0, half8);
      acc[t] = wmma_bf16(a, bfr, acc[t]);
    }
  }
  int mbase = m0 + ((lane >> 4) & 1) * 8;
#pragma unroll
  for (int t = 0; t < 4; ++t) {
    int col = n0 + t * 16 + (lane & 15);
#pragma unroll
    for (int i = 0; i < 8; ++i) {
      int m = mbase + i;
      int b = m >> 10, l = m & (L_ - 1);
      out[((size_t)(b * C_ + col)) * L_ + l] = acc[t][i];   // NCHW store
    }
  }
}

// ---------------- launch ----------------
extern "C" void kernel_launch(void* const* d_in, const int* in_sizes, int n_in,
                              void* d_out, int out_size, void* d_ws, size_t ws_size,
                              hipStream_t stream) {
  (void)in_sizes; (void)n_in; (void)out_size; (void)ws_size;
  const float* x          = (const float*)d_in[0];
  const float* in_proj_w  = (const float*)d_in[1];
  const float* conv_w     = (const float*)d_in[2];
  const float* conv_b     = (const float*)d_in[3];
  const float* x_proj_w   = (const float*)d_in[4];
  const float* A_logs     = (const float*)d_in[5];
  const float* Ds         = (const float*)d_in[6];
  const float* dt_projs_w = (const float*)d_in[7];
  const float* dt_projs_b = (const float*)d_in[8];
  const float* ln_gamma   = (const float*)d_in[9];
  const float* ln_beta    = (const float*)d_in[10];
  const float* out_proj_w = (const float*)d_in[11];
  float* out = (float*)d_out;

  char* ws = (char*)d_ws;
  size_t off = 0;
  auto alloc = [&](size_t bytes) -> char* {
    char* p = ws + off;
    off += (bytes + 255) & ~(size_t)255;
    return p;
  };

  bf16_t* xT_bf   = (bf16_t*)alloc((size_t)M1_ * C_ * 2);
  bf16_t* w_in_bf = (bf16_t*)alloc((size_t)2 * DI_ * C_ * 2);
  bf16_t* w80_bf  = (bf16_t*)alloc((size_t)(R_ + 2 * N_) * DI_ * 2);
  bf16_t* dtw_bf  = (bf16_t*)alloc((size_t)K_ * DI_ * RPAD * 2);
  bf16_t* outw_bf = (bf16_t*)alloc((size_t)C_ * DI_ * 2);
  float*  xi      = (float*)alloc((size_t)M1_ * DI_ * 4);
  float*  z_silu  = (float*)alloc((size_t)M1_ * DI_ * 4);
  float*  xcT     = (float*)alloc((size_t)M1_ * DI_ * 4);
  bf16_t* xcT_bf  = (bf16_t*)alloc((size_t)M1_ * DI_ * 2);
  bf16_t* dts_in  = (bf16_t*)alloc((size_t)B_ * K_ * L_ * RPAD * 2);
  float*  Bsb     = (float*)alloc((size_t)B_ * K_ * L_ * N_ * 4);
  float*  Csb     = (float*)alloc((size_t)B_ * K_ * L_ * N_ * 4);
  float*  dts     = (float*)alloc((size_t)B_ * K_ * L_ * DI_ * 4);
  float*  ys      = (float*)alloc((size_t)B_ * K_ * L_ * DI_ * 4);
  bf16_t* yln_bf  = (bf16_t*)alloc((size_t)M1_ * DI_ * 2);

  // 1) packs
  pack_x_kernel<<<(M1_ * C_ + 255) / 256, 256, 0, stream>>>(x, xT_bf);
  f32_to_bf16_kernel<<<(2 * DI_ * C_ + 255) / 256, 256, 0, stream>>>(in_proj_w, w_in_bf, 2 * DI_ * C_);
  f32_to_bf16_kernel<<<((R_ + 2 * N_) * DI_ + 255) / 256, 256, 0, stream>>>(x_proj_w, w80_bf, (R_ + 2 * N_) * DI_);
  f32_to_bf16_kernel<<<(C_ * DI_ + 255) / 256, 256, 0, stream>>>(out_proj_w, outw_bf, C_ * DI_);
  pack_dtw_kernel<<<(K_ * DI_ * RPAD + 255) / 256, 256, 0, stream>>>(dt_projs_w, dtw_bf);

  // 2) in-proj GEMM (WMMA) with xi / silu(z) split
  gemm_inproj<<<((M1_ / 16) * ((2 * DI_) / 64) * 32) / 256, 256, 0, stream>>>(xT_bf, w_in_bf, xi, z_silu);

  // 3) depthwise conv + silu
  conv_dw_kernel<<<(B_ * L_ * DI_ + 255) / 256, 256, 0, stream>>>(xi, conv_w, conv_b, xcT, xcT_bf);

  // 4) x_dbl GEMM (WMMA, gathered rows per direction) -> dts_in / Bs / Cs
  gemm_xdbl<<<(B_ * K_ * (L_ / 16) * 32) / 256, 256, 0, stream>>>(xcT_bf, w80_bf, dts_in, Bsb, Csb);

  // 5) dt-proj GEMM (WMMA)
  gemm_dts<<<(B_ * K_ * (L_ / 16) * (DI_ / 64) * 32) / 256, 256, 0, stream>>>(dts_in, dtw_bf, dts);

  // 6) selective scan
  scan_kernel<<<(B_ * K_ * DI_) / 128, 128, 0, stream>>>(dts, Bsb, Csb, xcT, A_logs, Ds, dt_projs_b, ys);

  // 7) merge directions + LayerNorm + gate
  merge_ln_kernel<<<B_ * L_, 256, 0, stream>>>(ys, z_silu, ln_gamma, ln_beta, yln_bf);

  // 8) out-proj GEMM (WMMA) with NCHW scatter store
  gemm_out<<<((M1_ / 16) * (C_ / 64) * 32) / 256, 256, 0, stream>>>(yln_bf, outw_bf, out);
}
